// CascadedHyperGAT_12266426597835
// MI455X (gfx1250) — matrix operand order repaired and checked by reading, not compiled, not run on hardware
//
#include <hip/hip_runtime.h>
#include <hip/hip_bf16.h>
#include <math.h>

#define N_NODES 100000
#define N_EDGES 50000
#define N_PAIRS 1000000
#define IN_DIM  256
#define OUT_DIM 128
#define TQ_DIM  64
#define HID     32
#define CAT_DIM 320
#define NEG_SLOPE 0.2f

typedef __attribute__((ext_vector_type(16))) __bf16 v16bf;
typedef __attribute__((ext_vector_type(8)))  float  v8f;

// ---------- monotone float <-> uint mapping for atomicMax on floats ----------
__device__ __forceinline__ unsigned f2ord(float f) {
  unsigned u = __float_as_uint(f);
  return (u & 0x80000000u) ? ~u : (u | 0x80000000u);
}
__device__ __forceinline__ float ord2f(unsigned u) {
  unsigned v = (u & 0x80000000u) ? (u & 0x7FFFFFFFu) : ~u;
  return __uint_as_float(v);
}

// ---------- init: zero outputs + denom, seed segment-max with ord(-inf) -----
__global__ void k_init(float* __restrict__ out, int out_n,
                       float* __restrict__ denom, unsigned* __restrict__ segmax) {
  int i = blockIdx.x * blockDim.x + threadIdx.x;
  if (i < out_n) out[i] = 0.0f;
  if (i < N_EDGES) { denom[i] = 0.0f; segmax[i] = f2ord(-INFINITY); }
}

// ---------- convert weights fp32 -> bf16 (contiguous row-major) -------------
__global__ void k_cvt(const float* __restrict__ Wq, const float* __restrict__ Wv,
                      const float* __restrict__ fc1,
                      __bf16* __restrict__ Wq_bf, __bf16* __restrict__ Wv_bf,
                      __bf16* __restrict__ fc1_bf) {
  int i = blockIdx.x * blockDim.x + threadIdx.x;
  if (i < OUT_DIM * IN_DIM) { Wq_bf[i] = (__bf16)Wq[i]; Wv_bf[i] = (__bf16)Wv[i]; }
  if (i < HID * CAT_DIM)    { fc1_bf[i] = (__bf16)fc1[i]; }
}

// ---------- kernel 1: per 16-node tile -> Q, V (WMMA bf16) + gated MLP ------
// A-fragment layout (16-bit A 16x32): lane m=lane&15 is row M,
//   group g=lane>>4; elem e -> K = 16*g + e (contiguous order chosen here, not the ISA table order).
// B-fragment layout (32x16): lane m is column N, elem e -> K = 16*g + e (same order as A).
// C/D layout: lane m = N, vgpr v -> M = v + 8*g.
__global__ void __launch_bounds__(256)
k_node_qv_mlp(const float* __restrict__ x, const int* __restrict__ node_types,
              const float* __restrict__ tq, const __bf16* __restrict__ fc1_bf,
              const float* __restrict__ fc1_b, const float* __restrict__ fc2_w,
              const float* __restrict__ fc2_b,
              const __bf16* __restrict__ Wq_bf, const __bf16* __restrict__ Wv_bf,
              float* __restrict__ Q, float* __restrict__ V,
              float* __restrict__ alpha_type) {
  const int lane = threadIdx.x & 31;
  const int tile = blockIdx.x * 8 + (threadIdx.x >> 5);
  if (tile >= N_NODES / 16) return;           // wave-uniform: EXEC stays all-1s
  const int r0 = tile * 16;
  const int m  = lane & 15;
  const int g  = lane >> 4;
  const int kb = g * 8;
  const int row = r0 + m;

  // cache x-tile as bf16 A fragments: 8 K-steps of 32
  v16bf A[8];
  const float* xr = x + (long)row * IN_DIM;
  for (int ks = 0; ks < 8; ++ks) {
    int k0 = ks * 32;
#pragma unroll
    for (int e = 0; e < 8; ++e) A[ks][e]     = (__bf16)xr[k0 + g * 16 + e];
#pragma unroll
    for (int e = 0; e < 8; ++e) A[ks][8 + e] = (__bf16)xr[k0 + g * 16 + 8 + e];
  }

  // ---- Q = x@Wq^T, V = x@Wv^T : 8 output col-tiles of 16 ----
  for (int nt = 0; nt < 8; ++nt) {
    int n0 = nt * 16;
    v8f accq = {}; v8f accv = {};
    const __bf16* wq = Wq_bf + (long)(n0 + m) * IN_DIM + g * 16;
    const __bf16* wv = Wv_bf + (long)(n0 + m) * IN_DIM + g * 16;
#pragma unroll
    for (int ks = 0; ks < 8; ++ks) {
      v16bf bq = *(const v16bf*)(wq + ks * 32);
      v16bf bv = *(const v16bf*)(wv + ks * 32);
      accq = __builtin_amdgcn_wmma_f32_16x16x32_bf16(false, A[ks], false, bq,
                                                     (short)0, accq, false, false);
      accv = __builtin_amdgcn_wmma_f32_16x16x32_bf16(false, A[ks], false, bv,
                                                     (short)0, accv, false, false);
    }
    float* qo = Q + (long)(r0 + g * 8) * OUT_DIM + n0 + m;
    float* vo = V + (long)(r0 + g * 8) * OUT_DIM + n0 + m;
#pragma unroll
    for (int v = 0; v < 8; ++v) { qo[v * OUT_DIM] = accq[v]; vo[v * OUT_DIM] = accv[v]; }
  }

  // ---- MLP: h = tanh([x, type_query[type]] @ fc1^T + b1) ----
  v16bf At[2];                                // K = 256..319 block from type_query
  const int trow = node_types[row];
  const float* tqr = tq + trow * TQ_DIM;
  for (int ks = 0; ks < 2; ++ks) {
    int k0 = ks * 32;
#pragma unroll
    for (int e = 0; e < 8; ++e) At[ks][e]     = (__bf16)tqr[k0 + g * 16 + e];
#pragma unroll
    for (int e = 0; e < 8; ++e) At[ks][8 + e] = (__bf16)tqr[k0 + g * 16 + 8 + e];
  }

  float h[2][8];
  for (int nt = 0; nt < 2; ++nt) {
    int n0 = nt * 16;
    v8f acc = {};
    const __bf16* w1 = fc1_bf + (long)(n0 + m) * CAT_DIM + g * 16;
#pragma unroll
    for (int ks = 0; ks < 8; ++ks) {
      v16bf b = *(const v16bf*)(w1 + ks * 32);
      acc = __builtin_amdgcn_wmma_f32_16x16x32_bf16(false, A[ks], false, b,
                                                    (short)0, acc, false, false);
    }
#pragma unroll
    for (int ks = 0; ks < 2; ++ks) {
      v16bf b = *(const v16bf*)(w1 + 256 + ks * 32);
      acc = __builtin_amdgcn_wmma_f32_16x16x32_bf16(false, At[ks], false, b,
                                                    (short)0, acc, false, false);
    }
    float bias = fc1_b[n0 + m];
#pragma unroll
    for (int v = 0; v < 8; ++v) h[nt][v] = tanhf(acc[v] + bias);
  }

  // ---- fc2 (32 -> 1) via cross-lane reduction over N (lanes within group) ----
  float w0 = fc2_w[m], w1c = fc2_w[16 + m];
  float p[8];
#pragma unroll
  for (int v = 0; v < 8; ++v) p[v] = h[0][v] * w0 + h[1][v] * w1c;
#pragma unroll
  for (int off = 1; off < 16; off <<= 1) {
#pragma unroll
    for (int v = 0; v < 8; ++v) p[v] += __shfl_xor(p[v], off, 32);
  }
  if (m == 0) {
    float b2 = fc2_b[0];
#pragma unroll
    for (int v = 0; v < 8; ++v) {
      float s = p[v] + b2;
      alpha_type[r0 + g * 8 + v] = 1.0f / (1.0f + __expf(-s));
    }
  }
}

// ---------- kernel 2: per-pair score + segment max (wave per pair) ----------
__global__ void __launch_bounds__(256)
k_pair_score(const float* __restrict__ Q, const float* __restrict__ alpha_type,
             const float* __restrict__ edge_ctx, const int* __restrict__ node_idx,
             const int* __restrict__ edge_idx, const int* __restrict__ edge_type,
             float* __restrict__ alpha, unsigned* __restrict__ segmax) {
  const int lane = threadIdx.x & 31;
  const long p = (long)blockIdx.x * 8 + (threadIdx.x >> 5);
  if (p >= N_PAIRS) return;
  const int n  = node_idx[p];
  const int e  = edge_idx[p];
  const int et = edge_type[e];
  const float4 q = *(const float4*)(Q + (long)n * OUT_DIM + lane * 4);
  const float4 c = *(const float4*)(edge_ctx + (long)et * OUT_DIM + lane * 4);
  float s = q.x * c.x + q.y * c.y + q.z * c.z + q.w * c.w;
#pragma unroll
  for (int off = 1; off < 32; off <<= 1) s += __shfl_xor(s, off, 32);
  if (lane == 0) {
    float an = (s >= 0.0f) ? s : NEG_SLOPE * s;   // leaky relu
    float a  = an * alpha_type[n];
    alpha[p] = a;
    atomicMax(segmax + e, f2ord(a));
  }
}

// ---------- kernel 3: exp(alpha - segmax) + denom scatter -------------------
__global__ void k_exp_denom(const float* __restrict__ alpha,
                            const int* __restrict__ edge_idx,
                            const unsigned* __restrict__ segmax,
                            float* __restrict__ ea, float* __restrict__ denom) {
  long p = (long)blockIdx.x * blockDim.x + threadIdx.x;
  if (p >= N_PAIRS) return;
  int e = edge_idx[p];
  float v = __expf(alpha[p] - ord2f(segmax[e]));
  ea[p] = v;
  atomicAdd(denom + e, v);
}

// ---------- kernel 4: edge_feat += attn * V[node]  (wave per pair) ----------
__global__ void __launch_bounds__(256)
k_edge_agg(const float* __restrict__ V, const float* __restrict__ ea,
           const float* __restrict__ denom, const int* __restrict__ node_idx,
           const int* __restrict__ edge_idx, float* __restrict__ edge_feat) {
  const int lane = threadIdx.x & 31;
  const long p = (long)blockIdx.x * 8 + (threadIdx.x >> 5);
  if (p >= N_PAIRS) return;
  const int n = node_idx[p];
  const int e = edge_idx[p];
  const float attn = ea[p] / denom[e];
  const float4 v = *(const float4*)(V + (long)n * OUT_DIM + lane * 4);
  float* dst = edge_feat + (long)e * OUT_DIM + lane * 4;
  atomicAdd(dst + 0, attn * v.x);
  atomicAdd(dst + 1, attn * v.y);
  atomicAdd(dst + 2, attn * v.z);
  atomicAdd(dst + 3, attn * v.w);
}

// ---------- kernel 5: node_feat += attn * edge_feat[edge] (wave per pair) ---
__global__ void __launch_bounds__(256)
k_node_agg(const float* __restrict__ ea, const float* __restrict__ denom,
           const int* __restrict__ node_idx, const int* __restrict__ edge_idx,
           const float* __restrict__ edge_feat, float* __restrict__ node_feat) {
  const int lane = threadIdx.x & 31;
  const long p = (long)blockIdx.x * 8 + (threadIdx.x >> 5);
  if (p >= N_PAIRS) return;
  const int n = node_idx[p];
  const int e = edge_idx[p];
  const float attn = ea[p] / denom[e];
  const float4 f = *(const float4*)(edge_feat + (long)e * OUT_DIM + lane * 4);
  float* dst = node_feat + (long)n * OUT_DIM + lane * 4;
  atomicAdd(dst + 0, attn * f.x);
  atomicAdd(dst + 1, attn * f.y);
  atomicAdd(dst + 2, attn * f.z);
  atomicAdd(dst + 3, attn * f.w);
}

extern "C" void kernel_launch(void* const* d_in, const int* in_sizes, int n_in,
                              void* d_out, int out_size, void* d_ws, size_t ws_size,
                              hipStream_t stream) {
  (void)in_sizes; (void)n_in; (void)out_size; (void)ws_size;

  const float* x          = (const float*)d_in[0];
  const int*   node_types = (const int*)  d_in[1];
  const int*   edge_type  = (const int*)  d_in[2];
  const int*   node_idx   = (const int*)  d_in[3];
  const int*   edge_idx   = (const int*)  d_in[4];
  const float* type_query = (const float*)d_in[5];
  const float* fc1_w      = (const float*)d_in[6];
  const float* fc1_b      = (const float*)d_in[7];
  const float* fc2_w      = (const float*)d_in[8];
  const float* fc2_b      = (const float*)d_in[9];
  const float* Wq         = (const float*)d_in[10];
  const float* Wv         = (const float*)d_in[11];
  const float* edge_ctx   = (const float*)d_in[12];

  float* out       = (float*)d_out;
  float* node_feat = out;                                   // (N, 128)
  float* edge_feat = out + (long)N_NODES * OUT_DIM;         // (E, 128)

  // workspace carve-out (256B aligned)
  char* ws = (char*)d_ws;
  size_t off = 0;
  auto alloc = [&](size_t bytes) -> char* {
    char* pp = ws + off;
    off = (off + bytes + 255) & ~(size_t)255;
    return pp;
  };
  float*    Qb         = (float*)   alloc((size_t)N_NODES * OUT_DIM * 4);
  float*    Vb         = (float*)   alloc((size_t)N_NODES * OUT_DIM * 4);
  float*    alpha_type = (float*)   alloc((size_t)N_NODES * 4);
  float*    alpha      = (float*)   alloc((size_t)N_PAIRS * 4);
  float*    eaw        = (float*)   alloc((size_t)N_PAIRS * 4);
  float*    denom      = (float*)   alloc((size_t)N_EDGES * 4);
  unsigned* segmax     = (unsigned*)alloc((size_t)N_EDGES * 4);
  __bf16*   Wq_bf      = (__bf16*)  alloc((size_t)OUT_DIM * IN_DIM * 2);
  __bf16*   Wv_bf      = (__bf16*)  alloc((size_t)OUT_DIM * IN_DIM * 2);
  __bf16*   fc1_bf     = (__bf16*)  alloc((size_t)HID * CAT_DIM * 2);

  const int tot = (N_NODES + N_EDGES) * OUT_DIM;
  k_init<<<(tot + 255) / 256, 256, 0, stream>>>(out, tot, denom, segmax);
  k_cvt<<<(OUT_DIM * IN_DIM + 255) / 256, 256, 0, stream>>>(Wq, Wv, fc1_w,
                                                            Wq_bf, Wv_bf, fc1_bf);
  k_node_qv_mlp<<<(N_NODES / 16 + 7) / 8, 256, 0, stream>>>(
      x, node_types, type_query, fc1_bf, fc1_b, fc2_w, fc2_b,
      Wq_bf, Wv_bf, Qb, Vb, alpha_type);
  k_pair_score<<<(N_PAIRS + 7) / 8, 256, 0, stream>>>(
      Qb, alpha_type, edge_ctx, node_idx, edge_idx, edge_type, alpha, segmax);
  k_exp_denom<<<(N_PAIRS + 255) / 256, 256, 0, stream>>>(
      alpha, edge_idx, segmax, eaw, denom);
  k_edge_agg<<<(N_PAIRS + 7) / 8, 256, 0, stream>>>(
      Vb, eaw, denom, node_idx, edge_idx, edge_feat);
  k_node_agg<<<(N_PAIRS + 7) / 8, 256, 0, stream>>>(
      eaw, denom, node_idx, edge_idx, edge_feat, node_feat);
}
